// LocalModel_11338713662113
// MI455X (gfx1250) — compile-verified, matrix operand order checked by reading, not run on hardware
//
#include <hip/hip_runtime.h>

// ---------------------------------------------------------------------------
// DeepGCN forward for MI455X (gfx1250, wave32, WMMA).
// H = 128 channels, DEPTH = 9.  GEMMs run as bf16 WMMA (f32 accumulate);
// edge aggregation uses f32 global atomics; BN in f32.
// ---------------------------------------------------------------------------

typedef __attribute__((ext_vector_type(16))) __bf16        v16bf;
typedef __attribute__((ext_vector_type(8)))  float         v8f;
typedef __attribute__((ext_vector_type(4)))  unsigned int  uv4;
typedef __attribute__((ext_vector_type(4)))  float         fv4;

union BFrag {
    v16bf v;
    uv4   q[2];
};

#define HDIM 128
#define BN_EPS 1e-5f

// ---------------------------------------------------------------------------
// Weight swizzle: f32 [128x128] row-major (k, c) -> bf16 WMMA B-fragments.
// Per layer layout: [ct(8)][ks(4)][lane(32)][e(16)] bf16, 16384 elements.
// Fragment index mapping matches the 16-bit 16x32 A-layout transposed onto B:
//   half = lane>>4, n = lane&15, c = ct*16 + n
//   e<8 : k = ks*32 + 8*half + e
//   e>=8: k = ks*32 + 16 + 8*half + (e-8)
// ---------------------------------------------------------------------------
__global__ void k_wswizzle(const float* __restrict__ Ws1,   // [9][128][128]
                           const float* __restrict__ Ws2,   // [8][128][128]
                           __bf16* __restrict__ wf)         // [17][16384]
{
    int t = blockIdx.x * 256 + threadIdx.x;
    int layer = t >> 14;
    if (layer >= 17) return;
    int idx  = t & 16383;
    const float* src = (layer < 9) ? (Ws1 + (size_t)layer * 16384)
                                   : (Ws2 + (size_t)(layer - 9) * 16384);
    int ct   = idx >> 11;
    int ks   = (idx >> 9) & 3;
    int lane = (idx >> 4) & 31;
    int e    = idx & 15;
    int half = lane >> 4;
    int n    = lane & 15;
    int c    = ct * 16 + n;
    int k    = ks * 32 + ((e < 8) ? (8 * half + e) : (16 + 8 * half + (e - 8)));
    wf[(size_t)t] = (__bf16)src[k * HDIM + c];
}

// ---------------------------------------------------------------------------
// GCN normalization
// ---------------------------------------------------------------------------
__global__ void k_fill1(float* __restrict__ p, int n)
{
    int i = blockIdx.x * 256 + threadIdx.x;
    if (i < n) p[i] = 1.0f;   // self loop contributes 1 to every degree
}

__global__ void k_deg_accum(const long long* __restrict__ dst,
                            float* __restrict__ deg, long long E)
{
    long long e = (long long)blockIdx.x * 256 + threadIdx.x;
    if (e < E) atomicAdd(&deg[dst[e]], 1.0f);
}

__global__ void k_dinv(float* __restrict__ deg, int n)
{
    int i = blockIdx.x * 256 + threadIdx.x;
    if (i < n) deg[i] = rsqrtf(deg[i]);   // deg >= 1 always (self loops)
}

__global__ void k_norm(const long long* __restrict__ src,
                       const long long* __restrict__ dst,
                       const float* __restrict__ dinv,
                       float* __restrict__ nrm, long long E)
{
    long long e = (long long)blockIdx.x * 256 + threadIdx.x;
    if (e < E) nrm[e] = dinv[src[e]] * dinv[dst[e]];
}

// ---------------------------------------------------------------------------
// First layer: h = x[N,3] @ W0[3,128]  (K=3, plain VALU)
// ---------------------------------------------------------------------------
__global__ void k_first_layer(const float* __restrict__ x,
                              const float* __restrict__ W0,
                              float* __restrict__ h, int N)
{
    int t = blockIdx.x * 256 + threadIdx.x;
    if (t >= N * HDIM) return;
    int i = t >> 7;
    int c = t & 127;
    const float* xr = x + (size_t)i * 3;
    h[t] = xr[0] * W0[c] + xr[1] * W0[HDIM + c] + xr[2] * W0[2 * HDIM + c];
}

// ---------------------------------------------------------------------------
// Main GEMM: C[N,128] = A[N,128](bf16) @ W(bf16 fragments), f32 accumulate.
// One wave per 16x16 tile; 8 waves/block cover all 8 column tiles of 16 rows.
// K = 128 -> 4x v_wmma_f32_16x16x32_bf16.
// ---------------------------------------------------------------------------
__global__ void k_gemm_wmma(const __bf16* __restrict__ A,   // [N,128] row-major
                            const __bf16* __restrict__ Wf,  // [8][4][32][16] frags
                            float* __restrict__ C,          // [N,128]
                            int N)
{
    const int wave = threadIdx.x >> 5;      // column tile 0..7
    const int lane = threadIdx.x & 31;
    const int half = lane >> 4;
    const int l15  = lane & 15;
    const long long row0 = (long long)blockIdx.x * 16;
    if (row0 + 16 > N) return;              // uniform guard keeps EXEC all-ones

    const __bf16* arow = A + (row0 + l15) * HDIM;

    v8f acc = {};
    #pragma unroll
    for (int ks = 0; ks < 4; ++ks) {
        const int kk = ks * 32;
        BFrag a, b;
        a.q[0] = *(const uv4*)(arow + kk + 8 * half);
        a.q[1] = *(const uv4*)(arow + kk + 16 + 8 * half);
        const __bf16* bp = Wf + ((size_t)(wave * 4 + ks) * 32 + lane) * 16;
        b.q[0] = *(const uv4*)(bp);
        b.q[1] = *(const uv4*)(bp + 8);
        acc = __builtin_amdgcn_wmma_f32_16x16x32_bf16(
                  false, a.v, false, b.v, (short)0, acc, false, false);
    }

    float* cbase = C + (row0 + 8 * half) * HDIM + wave * 16 + l15;
    #pragma unroll
    for (int r = 0; r < 8; ++r)
        cbase[(size_t)r * HDIM] = acc[r];
}

// ---------------------------------------------------------------------------
// Aggregation: agg[i,:] = dinv[i]^2 * h[i,:]  then  agg[dst,:] += norm_e*h[src,:]
// ---------------------------------------------------------------------------
__global__ void k_self_init(const float* __restrict__ h,
                            const float* __restrict__ dinv,
                            float* __restrict__ agg, int N)
{
    int t = blockIdx.x * 256 + threadIdx.x;
    if (t >= N * HDIM) return;
    float di = dinv[t >> 7];
    agg[t] = di * di * h[t];
}

__global__ void k_edge_scatter(const float* __restrict__ h,
                               const long long* __restrict__ src,
                               const long long* __restrict__ dst,
                               const float* __restrict__ nrm,
                               float* __restrict__ agg, long long E)
{
    long long e = (long long)blockIdx.x * 8 + (threadIdx.x >> 5);
    if (e >= E) return;
    const int lane = threadIdx.x & 31;
    const long long s = src[e];
    const long long d = dst[e];
    const float w = nrm[e];
    const fv4 hv = *(const fv4*)(h + s * HDIM + lane * 4);
    float* ap = agg + d * HDIM + lane * 4;
    atomicAdd(ap + 0, w * hv.x);
    atomicAdd(ap + 1, w * hv.y);
    atomicAdd(ap + 2, w * hv.z);
    atomicAdd(ap + 3, w * hv.w);
}

// ---------------------------------------------------------------------------
// BatchNorm (training batch stats) + ReLU -> bf16 activation
// ---------------------------------------------------------------------------
__global__ void k_zero256(float* __restrict__ p)
{
    p[threadIdx.x] = 0.0f;
}

__global__ void k_bn_stats(const float* __restrict__ x,
                           float* __restrict__ stats,  // [0..127]=sum [128..255]=sumsq
                           int N)
{
    __shared__ float ssum[256];
    __shared__ float ssq[256];
    const int c    = threadIdx.x & 127;
    const int half = threadIdx.x >> 7;
    float s = 0.f, q = 0.f;
    for (int r = blockIdx.x * 2 + half; r < N; r += gridDim.x * 2) {
        float v = x[(size_t)r * HDIM + c];
        s += v;
        q += v * v;
    }
    ssum[threadIdx.x] = s;
    ssq[threadIdx.x]  = q;
    __syncthreads();
    if (threadIdx.x < 128) {
        atomicAdd(&stats[c],        ssum[threadIdx.x] + ssum[threadIdx.x + 128]);
        atomicAdd(&stats[128 + c],  ssq[threadIdx.x]  + ssq[threadIdx.x + 128]);
    }
}

__global__ void k_bn_relu_apply(const float* __restrict__ x,
                                const float* __restrict__ stats,
                                const float* __restrict__ gamma,
                                const float* __restrict__ beta,
                                __bf16* __restrict__ out, int N)
{
    int t = blockIdx.x * 256 + threadIdx.x;
    if (t >= N * HDIM) return;
    int c = t & 127;
    float invN = 1.0f / (float)N;
    float m   = stats[c] * invN;
    float var = stats[128 + c] * invN - m * m;
    float is  = rsqrtf(var + BN_EPS);
    float y   = (x[t] - m) * is * gamma[c] + beta[c];
    out[t] = (__bf16)fmaxf(y, 0.0f);
}

__global__ void k_skip_add(const __bf16* __restrict__ a,
                           const __bf16* __restrict__ b,
                           __bf16* __restrict__ out, int N)
{
    int t = blockIdx.x * 256 + threadIdx.x;
    if (t < N * HDIM) out[t] = (__bf16)((float)a[t] + (float)b[t]);
}

// ---------------------------------------------------------------------------
// Final 128 -> 1 projection, scalar aggregation, sigmoid
// ---------------------------------------------------------------------------
__global__ void k_final_dot(const __bf16* __restrict__ a,  // [N,128]
                            const float* __restrict__ Wout,
                            float* __restrict__ z, int N)
{
    int i = blockIdx.x * 8 + (threadIdx.x >> 5);
    if (i >= N) return;
    const int lane = threadIdx.x & 31;
    const __bf16* ar = a + (size_t)i * HDIM + lane * 4;
    float s = (float)ar[0] * Wout[lane * 4 + 0] +
              (float)ar[1] * Wout[lane * 4 + 1] +
              (float)ar[2] * Wout[lane * 4 + 2] +
              (float)ar[3] * Wout[lane * 4 + 3];
    #pragma unroll
    for (int off = 16; off > 0; off >>= 1)
        s += __shfl_down(s, off, 32);
    if (lane == 0) z[i] = s;
}

__global__ void k_final_self(const float* __restrict__ z,
                             const float* __restrict__ dinv,
                             float* __restrict__ aggs, int N)
{
    int i = blockIdx.x * 256 + threadIdx.x;
    if (i < N) {
        float di = dinv[i];
        aggs[i] = di * di * z[i];
    }
}

__global__ void k_final_scatter(const float* __restrict__ z,
                                const long long* __restrict__ src,
                                const long long* __restrict__ dst,
                                const float* __restrict__ nrm,
                                float* __restrict__ aggs, long long E)
{
    long long e = (long long)blockIdx.x * 256 + threadIdx.x;
    if (e < E) atomicAdd(&aggs[dst[e]], nrm[e] * z[src[e]]);
}

__global__ void k_sigmoid(const float* __restrict__ aggs,
                          float* __restrict__ out, int N)
{
    int i = blockIdx.x * 256 + threadIdx.x;
    if (i < N) out[i] = 1.0f / (1.0f + expf(-aggs[i]));
}

// ---------------------------------------------------------------------------
// Host-side layer helpers
// ---------------------------------------------------------------------------
static void run_aggregate_bn(const float* hbuf, float* aggbuf, float* stats,
                             const float* dinv,
                             const long long* srcI, const long long* dstI,
                             const float* nrm,
                             const float* gamma, const float* beta,
                             __bf16* out, int N, long long E, hipStream_t st)
{
    int nc = N * HDIM;
    k_self_init<<<(nc + 255) / 256, 256, 0, st>>>(hbuf, dinv, aggbuf, N);
    k_edge_scatter<<<(unsigned)((E + 7) / 8), 256, 0, st>>>(hbuf, srcI, dstI, nrm, aggbuf, E);
    k_zero256<<<1, 256, 0, st>>>(stats);
    k_bn_stats<<<1024, 256, 0, st>>>(aggbuf, stats, N);
    k_bn_relu_apply<<<(nc + 255) / 256, 256, 0, st>>>(aggbuf, stats, gamma, beta, out, N);
}

extern "C" void kernel_launch(void* const* d_in, const int* in_sizes, int n_in,
                              void* d_out, int out_size, void* d_ws, size_t ws_size,
                              hipStream_t stream)
{
    const float*     x    = (const float*)d_in[0];
    const long long* ei   = (const long long*)d_in[1];   // int64 per reference
    const float*     W0   = (const float*)d_in[2];
    const float*     Ws1  = (const float*)d_in[3];
    const float*     g1   = (const float*)d_in[4];
    const float*     b1   = (const float*)d_in[5];
    const float*     Ws2  = (const float*)d_in[6];
    const float*     g2   = (const float*)d_in[7];
    const float*     b2   = (const float*)d_in[8];
    const float*     Wout = (const float*)d_in[9];

    const int       N = in_sizes[0] / 3;
    const long long E = in_sizes[1] / 2;
    const long long* srcI = ei;
    const long long* dstI = ei + E;
    const size_t NC = (size_t)N * HDIM;

    // ---- carve workspace -------------------------------------------------
    char* p = (char*)d_ws;
    auto carve = [&](size_t bytes) -> void* {
        void* r = (void*)p;
        p += (bytes + 255) & ~(size_t)255;
        return r;
    };
    __bf16* wfrag  = (__bf16*)carve((size_t)17 * 16384 * sizeof(__bf16));
    float*  dinv   = (float*) carve((size_t)N * sizeof(float));      // deg -> dinv
    float*  nrm    = (float*) carve((size_t)E * sizeof(float));
    float*  hbuf   = (float*) carve(NC * sizeof(float));
    float*  aggbuf = (float*) carve(NC * sizeof(float));
    float*  stats  = (float*) carve(256 * sizeof(float));
    __bf16* xs     = (__bf16*)carve(9 * NC * sizeof(__bf16));        // saved skips
    __bf16* curA   = (__bf16*)carve(NC * sizeof(__bf16));
    __bf16* curB   = (__bf16*)carve(NC * sizeof(__bf16));
    float*  zscal  = (float*) carve((size_t)N * sizeof(float));
    float*  aggs   = (float*) carve((size_t)N * sizeof(float));
    (void)ws_size; (void)n_in;

    const int nc      = (int)NC;
    const int gElem   = (nc + 255) / 256;
    const int gNode   = (N + 255) / 256;
    const unsigned gE = (unsigned)((E + 255) / 256);

    // ---- weights -> bf16 WMMA fragments ---------------------------------
    k_wswizzle<<<(17 * 16384) / 256, 256, 0, stream>>>(Ws1, Ws2, wfrag);

    // ---- GCN symmetric normalization ------------------------------------
    k_fill1<<<gNode, 256, 0, stream>>>(dinv, N);
    k_deg_accum<<<gE, 256, 0, stream>>>(dstI, dinv, E);
    k_dinv<<<gNode, 256, 0, stream>>>(dinv, N);
    k_norm<<<gE, 256, 0, stream>>>(srcI, dstI, dinv, nrm, E);

    // ---- conv1[0]: x @ W0, aggregate, BN+ReLU -> xs[0] ------------------
    k_first_layer<<<gElem, 256, 0, stream>>>(x, W0, hbuf, N);
    run_aggregate_bn(hbuf, aggbuf, stats, dinv, srcI, dstI, nrm,
                     g1, b1, xs, N, E, stream);

    // ---- down path: layers 1..9 (Ws1), save xs[1..8] --------------------
    const __bf16* cur = xs;                   // xs[0]
    for (int i = 1; i <= 9; ++i) {
        const __bf16* wl = wfrag + (size_t)(i - 1) * 16384;
        __bf16* out = (i < 9) ? (xs + (size_t)i * NC) : curA;
        k_gemm_wmma<<<N / 16, 256, 0, stream>>>(cur, wl, hbuf, N);
        run_aggregate_bn(hbuf, aggbuf, stats, dinv, srcI, dstI, nrm,
                         g1 + i * HDIM, b1 + i * HDIM, out, N, E, stream);
        cur = out;
    }

    // ---- up path: skips + layers (Ws2), final Wout + sigmoid ------------
    for (int i = 0; i < 9; ++i) {
        const int j = 8 - i;
        k_skip_add<<<gElem, 256, 0, stream>>>(cur, xs + (size_t)j * NC, curB, N);
        if (i < 8) {
            const __bf16* wl = wfrag + (size_t)(9 + i) * 16384;
            k_gemm_wmma<<<N / 16, 256, 0, stream>>>(curB, wl, hbuf, N);
            run_aggregate_bn(hbuf, aggbuf, stats, dinv, srcI, dstI, nrm,
                             g2 + i * HDIM, b2 + i * HDIM, curA, N, E, stream);
            cur = curA;
        } else {
            k_final_dot<<<(N + 7) / 8, 256, 0, stream>>>(curB, Wout, zscal, N);
            k_final_self<<<gNode, 256, 0, stream>>>(zscal, dinv, aggs, N);
            k_final_scatter<<<gE, 256, 0, stream>>>(zscal, srcI, dstI, nrm, aggs, E);
            k_sigmoid<<<gNode, 256, 0, stream>>>(aggs, (float*)d_out, N);
        }
    }
}
